// Attention_944892805701
// MI455X (gfx1250) — matrix operand order repaired and checked by reading, not compiled, not run on hardware
//
#include <hip/hip_runtime.h>
#include <hip/hip_bf16.h>
#include <stdint.h>

#define DIM   4096
#define SEQ   2048
#define NH    32
#define NKV   8
#define HD    128
#define SCALE 0.08838834764831845f  // 1/sqrt(128)

typedef unsigned short u16;
typedef __attribute__((ext_vector_type(16))) __bf16 v16bf;
typedef __attribute__((ext_vector_type(8)))  __bf16 v8bf;
typedef __attribute__((ext_vector_type(8)))  float  v8f;
typedef int v4i __attribute__((vector_size(16)));   // int4 vector for async builtin

#if defined(__has_builtin)
#  if __has_builtin(__builtin_amdgcn_global_load_async_to_lds_b128)
#    define HAVE_ASYNC_LDS 1
#  endif
#endif

// ---- helpers -------------------------------------------------------------

__device__ __forceinline__ u16 f2bf(float f) {        // hardware RNE cvt
    __bf16 h = (__bf16)f;
    return __builtin_bit_cast(u16, h);
}
__device__ __forceinline__ unsigned pack2(float a, float b) {
    return (unsigned)f2bf(a) | ((unsigned)f2bf(b) << 16);
}
// Build a v16bf fragment from two 16B LDS chunks (lowers to ds_load_b128 x2)
__device__ __forceinline__ v16bf load_frag(const u16* p0, const u16* p1) {
    v8bf lo = *(const v8bf*)p0;
    v8bf hi = *(const v8bf*)p1;
    return __builtin_shufflevector(lo, hi, 0,1,2,3,4,5,6,7,8,9,10,11,12,13,14,15);
}
// 16B global -> LDS copy: async DMA path when available
__device__ __forceinline__ void copy16_g2l(u16* lds, const u16* glb) {
#ifdef HAVE_ASYNC_LDS
    __builtin_amdgcn_global_load_async_to_lds_b128(
        (__attribute__((address_space(1))) v4i*)glb,
        (__attribute__((address_space(3))) v4i*)lds, 0, 0);
#else
    *(uint4*)lds = *(const uint4*)glb;
#endif
}
__device__ __forceinline__ void async_wait() {
#ifdef HAVE_ASYNC_LDS
#  if __has_builtin(__builtin_amdgcn_s_wait_asynccnt)
    __builtin_amdgcn_s_wait_asynccnt(0);
#  else
    asm volatile("s_wait_asynccnt 0x0" ::: "memory");
#  endif
#endif
}

// ---- kernel 1: QKV projection + RoPE ------------------------------------
// grid (32 mtiles, 48 ntiles): ntile 0..31 -> Q head, 32..39 -> K head, 40..47 -> V head
__global__ __launch_bounds__(256) void qkv_rope_kernel(
    const float* __restrict__ data, const float* __restrict__ Wq,
    const float* __restrict__ Wk,  const float* __restrict__ Wv,
    const float* __restrict__ cosT, const float* __restrict__ sinT,
    u16* __restrict__ Qr, u16* __restrict__ Kr, u16* __restrict__ VT)
{
    __shared__ u16  As[128 * 40];    // [m][k] bf16, stride 40 (80B rows)
    __shared__ u16  Bs[128 * 40];    // [n][k] bf16
    __shared__ float Ds[128 * 132];  // staged f32 output tile

    const int mtile = blockIdx.x;
    const int ntile = blockIdx.y;
    const int tid   = threadIdx.x;
    const int lane  = tid & 31;
    const int wave  = tid >> 5;         // 8 waves: 4 (M) x 2 (N)
    const int wm    = (wave >> 1) * 32;
    const int wn    = (wave & 1)  * 64;
    const int laneN = lane & 15;
    const int half  = lane >> 4;

    const float* W; int n0, kind, head;
    if (ntile < 32)      { W = Wq; n0 = ntile * 128;        kind = 0; head = ntile; }
    else if (ntile < 40) { W = Wk; n0 = (ntile - 32) * 128; kind = 1; head = ntile - 32; }
    else                 { W = Wv; n0 = (ntile - 40) * 128; kind = 2; head = ntile - 40; }

    const int gm0 = mtile * 128;

    v8f acc[2][4];
    #pragma unroll
    for (int a = 0; a < 2; ++a)
        #pragma unroll
        for (int t = 0; t < 4; ++t) { v8f z = {}; acc[a][t] = z; }

    for (int k0 = 0; k0 < DIM; k0 += 32) {
        __syncthreads();
        #pragma unroll
        for (int i = 0; i < 4; ++i) {          // 128x32 f32 tiles -> bf16 LDS
            int idx = i * 256 + tid;           // 0..1023
            int m   = idx >> 3;
            int kc  = (idx & 7) << 2;
            float4 v = *(const float4*)(data + (size_t)(gm0 + m) * DIM + k0 + kc);
            uint2 pa; pa.x = pack2(v.x, v.y); pa.y = pack2(v.z, v.w);
            *(uint2*)&As[m * 40 + kc] = pa;
            float4 w = *(const float4*)(W + (size_t)(n0 + m) * DIM + k0 + kc);
            uint2 pb; pb.x = pack2(w.x, w.y); pb.y = pack2(w.z, w.w);
            *(uint2*)&Bs[m * 40 + kc] = pb;
        }
        if (k0 + 32 < DIM) {                   // emits global_prefetch_b8
            __builtin_prefetch(data + (size_t)(gm0 + (tid >> 1)) * DIM + k0 + 32, 0, 0);
            __builtin_prefetch(W    + (size_t)(n0  + (tid >> 1)) * DIM + k0 + 32, 0, 0);
        }
        __syncthreads();

        v16bf af[2], bfq[4];
        #pragma unroll
        for (int a = 0; a < 2; ++a) {          // A 16x32: K = {8h..8h+7, 16+8h..}
            const u16* base = &As[(wm + a * 16 + laneN) * 40 + half * 8];
            af[a] = load_frag(base, base + 16);
        }
        #pragma unroll
        for (int t = 0; t < 4; ++t) {          // B 32x16: K = {8h..8h+7, 16+8h..} for col n
            const u16* base = &Bs[(wn + t * 16 + laneN) * 40 + half * 8];
            bfq[t] = load_frag(base, base + 16);
        }
        #pragma unroll
        for (int a = 0; a < 2; ++a)
            #pragma unroll
            for (int t = 0; t < 4; ++t)
                acc[a][t] = __builtin_amdgcn_wmma_f32_16x16x32_bf16(
                    false, af[a], false, bfq[t], (short)0, acc[a][t], false, false);
    }

    // stage D tile to LDS so RoPE can pair columns n and n±64 across waves
    #pragma unroll
    for (int a = 0; a < 2; ++a)
        #pragma unroll
        for (int t = 0; t < 4; ++t)
            #pragma unroll
            for (int r = 0; r < 8; ++r)
                Ds[(wm + a * 16 + r + 8 * half) * 132 + (wn + t * 16 + laneN)] = acc[a][t][r];
    __syncthreads();

    if (kind == 2) {                            // V: store transposed [d][s] bf16
        for (int i = 0; i < 64; ++i) {
            int e = i * 256 + tid;
            int m = e & 127, n = e >> 7;
            int gm = gm0 + m; int b = gm >> 11; int s = gm & (SEQ - 1);
            VT[((size_t)(b * NKV + head) * HD + n) * SEQ + s] = f2bf(Ds[m * 132 + n]);
        }
    } else {                                    // Q/K: RoPE then store [b][h][s][d]
        for (int i = 0; i < 64; ++i) {
            int e = i * 256 + tid;
            int n = e & 127, m = e >> 7;
            int gm = gm0 + m; int b = gm >> 11; int s = gm & (SEQ - 1);
            float v   = Ds[m * 132 + n];
            float rot = (n < 64) ? -Ds[m * 132 + n + 64] : Ds[m * 132 + n - 64];
            float o   = v * cosT[s * HD + n] + rot * sinT[s * HD + n];
            u16 ov = f2bf(o);
            if (kind == 0) Qr[((size_t)(b * NH  + head) * SEQ + s) * HD + n] = ov;
            else           Kr[((size_t)(b * NKV + head) * SEQ + s) * HD + n] = ov;
        }
    }
}

// ---- kernel 2: causal flash attention (GQA) ------------------------------
// grid (16 qtiles, 64 b*h)
__global__ __launch_bounds__(256) void attn_kernel(
    const u16* __restrict__ Qr, const u16* __restrict__ Kr,
    const u16* __restrict__ VT, u16* __restrict__ ctx)
{
    __shared__ u16 Qs[128 * 136];     // [m][d]
    __shared__ u16 Ks[128 * 136];     // [j][d]
    __shared__ u16 Vs[128 * 136];     // [d][j]  (from global V^T)
    __shared__ u16 Ps[8][16 * 136];   // per-wave P scratch [m][j]

    const int qtile = blockIdx.x;
    const int bh    = blockIdx.y;
    const int b = bh >> 5, h = bh & 31;
    const int hk = h >> 2;            // N_REP = 4
    const int tid = threadIdx.x, lane = tid & 31, wave = tid >> 5;
    const int laneN = lane & 15, half = lane >> 4;
    const int q0 = qtile * 128;

    const u16* Qg = Qr + ((size_t)(b * NH  + h)  * SEQ + q0) * HD;
    const u16* Kg = Kr + ((size_t)(b * NKV + hk) * SEQ) * HD;
    const u16* Vg = VT + ((size_t)(b * NKV + hk) * HD) * SEQ;

    #pragma unroll
    for (int i = 0; i < 8; ++i) {     // Q tile -> LDS (async b128 copies)
        int idx = i * 256 + tid;
        int m = idx >> 4, c = (idx & 15) * 8;
        copy16_g2l(&Qs[m * 136 + c], Qg + (size_t)m * HD + c);
    }
    async_wait();
    __syncthreads();

    v16bf qf[4];                      // Q A-frags resident for whole kernel
    #pragma unroll
    for (int ks = 0; ks < 4; ++ks) {
        const u16* base = &Qs[(wave * 16 + laneN) * 136 + ks * 32 + half * 8];
        qf[ks] = load_frag(base, base + 16);
    }

    float mrow[8], lrow[8];
    v8f oacc[8];
    #pragma unroll
    for (int r = 0; r < 8; ++r) { mrow[r] = -3.0e38f; lrow[r] = 0.f; }
    #pragma unroll
    for (int t = 0; t < 8; ++t) { v8f z = {}; oacc[t] = z; }

    for (int kb = 0; kb <= qtile; ++kb) {
        __syncthreads();
        const int j0 = kb * 128;
        #pragma unroll
        for (int i = 0; i < 8; ++i) {
            int idx = i * 256 + tid;
            int m = idx >> 4, c = (idx & 15) * 8;
            copy16_g2l(&Ks[m * 136 + c], Kg + (size_t)(j0 + m) * HD + c);
            copy16_g2l(&Vs[m * 136 + c], Vg + (size_t)m * SEQ + j0 + c);
        }
        async_wait();
        __syncthreads();

        // logits: S = Q K^T  (wave rows m0 = wave*16, 8 j-tiles)
        v8f st[8];
        #pragma unroll
        for (int t = 0; t < 8; ++t) {
            v8f c = {};
            #pragma unroll
            for (int ks = 0; ks < 4; ++ks) {
                const u16* base = &Ks[(t * 16 + laneN) * 136 + ks * 32 + half * 8];
                v16bf kf = load_frag(base, base + 16);
                c = __builtin_amdgcn_wmma_f32_16x16x32_bf16(
                        false, qf[ks], false, kf, (short)0, c, false, false);
            }
            st[t] = c;
        }

        const bool diag  = (kb == qtile);
        const int  qbase = q0 + wave * 16 + 8 * half;   // + r -> query index
        #pragma unroll
        for (int t = 0; t < 8; ++t) {
            int j = j0 + t * 16 + laneN;
            #pragma unroll
            for (int r = 0; r < 8; ++r) {
                float v = st[t][r] * SCALE;
                if (diag && j > qbase + r) v = -3.0e38f;
                st[t][r] = v;
            }
        }

        // online softmax per row (rows live across 16-lane groups)
        #pragma unroll
        for (int r = 0; r < 8; ++r) {
            float mx = st[0][r];
            #pragma unroll
            for (int t = 1; t < 8; ++t) mx = fmaxf(mx, st[t][r]);
            #pragma unroll
            for (int off = 1; off < 16; off <<= 1) mx = fmaxf(mx, __shfl_xor(mx, off, 32));
            float mnew  = fmaxf(mrow[r], mx);
            float alpha = __expf(mrow[r] - mnew);
            mrow[r] = mnew;
            lrow[r] *= alpha;
            #pragma unroll
            for (int t = 0; t < 8; ++t) oacc[t][r] *= alpha;
            float rs = 0.f;
            #pragma unroll
            for (int t = 0; t < 8; ++t) {
                float p = __expf(st[t][r] - mnew);
                st[t][r] = p;
                rs += p;
            }
            #pragma unroll
            for (int off = 1; off < 16; off <<= 1) rs += __shfl_xor(rs, off, 32);
            lrow[r] += rs;
        }

        // D-layout -> A-layout via per-wave LDS, then O += P V
        u16* pw = &Ps[wave][0];
        #pragma unroll
        for (int t = 0; t < 8; ++t)
            #pragma unroll
            for (int r = 0; r < 8; ++r)
                pw[(r + 8 * half) * 136 + t * 16 + laneN] = f2bf(st[t][r]);

        v16bf pf[4];
        #pragma unroll
        for (int ks = 0; ks < 4; ++ks) {
            const u16* base = &pw[laneN * 136 + ks * 32 + half * 8];
            pf[ks] = load_frag(base, base + 16);
        }
        #pragma unroll
        for (int t = 0; t < 8; ++t)
            #pragma unroll
            for (int ks = 0; ks < 4; ++ks) {
                const u16* base = &Vs[(t * 16 + laneN) * 136 + ks * 32 + half * 8];
                v16bf vf = load_frag(base, base + 16);
                oacc[t] = __builtin_amdgcn_wmma_f32_16x16x32_bf16(
                              false, pf[ks], false, vf, (short)0, oacc[t], false, false);
            }
    }

    // normalize + store ctx bf16 [b][s][h*128 + d]
    #pragma unroll
    for (int r = 0; r < 8; ++r) {
        float inv = 1.0f / lrow[r];
        #pragma unroll
        for (int t = 0; t < 8; ++t) oacc[t][r] *= inv;
    }
    u16* cg = ctx + ((size_t)b * SEQ + q0) * DIM + h * HD;
    #pragma unroll
    for (int t = 0; t < 8; ++t)
        #pragma unroll
        for (int r = 0; r < 8; ++r) {
            int m = wave * 16 + r + 8 * half;
            int d = t * 16 + laneN;
            cg[(size_t)m * DIM + d] = f2bf(oacc[t][r]);
        }
}

// ---- kernel 3: output projection ctx @ Wo^T -> f32 -----------------------
// grid (32 mtiles, 32 ntiles)
__global__ __launch_bounds__(256) void oproj_kernel(
    const u16* __restrict__ ctx, const float* __restrict__ Wo,
    float* __restrict__ out)
{
    __shared__ u16 As[128 * 40];
    __shared__ u16 Bs[128 * 40];

    const int mtile = blockIdx.x, ntile = blockIdx.y;
    const int tid = threadIdx.x, lane = tid & 31, wave = tid >> 5;
    const int wm = (wave >> 1) * 32, wn = (wave & 1) * 64;
    const int laneN = lane & 15, half = lane >> 4;
    const int gm0 = mtile * 128, n0 = ntile * 128;

    v8f acc[2][4];
    #pragma unroll
    for (int a = 0; a < 2; ++a)
        #pragma unroll
        for (int t = 0; t < 4; ++t) { v8f z = {}; acc[a][t] = z; }

    for (int k0 = 0; k0 < DIM; k0 += 32) {
        __syncthreads();
        #pragma unroll
        for (int i = 0; i < 2; ++i) {          // A already bf16: async b128 copies
            int idx = i * 256 + tid;
            int m = idx >> 2, c = (idx & 3) * 8;
            copy16_g2l(&As[m * 40 + c], ctx + (size_t)(gm0 + m) * DIM + k0 + c);
        }
        #pragma unroll
        for (int i = 0; i < 4; ++i) {          // B: f32 -> bf16
            int idx = i * 256 + tid;
            int m = idx >> 3, kc = (idx & 7) << 2;
            float4 w = *(const float4*)(Wo + (size_t)(n0 + m) * DIM + k0 + kc);
            uint2 pb; pb.x = pack2(w.x, w.y); pb.y = pack2(w.z, w.w);
            *(uint2*)&Bs[m * 40 + kc] = pb;
        }
        if (k0 + 32 < DIM)
            __builtin_prefetch(Wo + (size_t)(n0 + (tid >> 1)) * DIM + k0 + 32, 0, 0);
        async_wait();
        __syncthreads();

        v16bf af[2], bfq[4];
        #pragma unroll
        for (int a = 0; a < 2; ++a) {
            const u16* base = &As[(wm + a * 16 + laneN) * 40 + half * 8];
            af[a] = load_frag(base, base + 16);
        }
        #pragma unroll
        for (int t = 0; t < 4; ++t) {
            const u16* base = &Bs[(wn + t * 16 + laneN) * 40 + half * 8];
            bfq[t] = load_frag(base, base + 16);
        }
        #pragma unroll
        for (int a = 0; a < 2; ++a)
            #pragma unroll
            for (int t = 0; t < 4; ++t)
                acc[a][t] = __builtin_amdgcn_wmma_f32_16x16x32_bf16(
                    false, af[a], false, bfq[t], (short)0, acc[a][t], false, false);
    }

    #pragma unroll
    for (int a = 0; a < 2; ++a)
        #pragma unroll
        for (int t = 0; t < 4; ++t)
            #pragma unroll
            for (int r = 0; r < 8; ++r) {
                int m = gm0 + wm + a * 16 + r + 8 * half;
                int n = n0 + wn + t * 16 + laneN;
                out[(size_t)m * DIM + n] = acc[a][t][r];
            }
}

// ---- launch --------------------------------------------------------------

extern "C" void kernel_launch(void* const* d_in, const int* in_sizes, int n_in,
                              void* d_out, int out_size, void* d_ws, size_t ws_size,
                              hipStream_t stream) {
    const float* data = (const float*)d_in[0];
    const float* Wq   = (const float*)d_in[1];
    const float* Wk   = (const float*)d_in[2];
    const float* Wv   = (const float*)d_in[3];
    const float* Wo   = (const float*)d_in[4];
    const float* cosT = (const float*)d_in[5];
    const float* sinT = (const float*)d_in[6];
    float* out = (float*)d_out;

    u16* ws  = (u16*)d_ws;
    u16* Qr  = ws;                                   // 2*32*2048*128
    u16* Kr  = Qr  + (size_t)2 * NH  * SEQ * HD;     // 2*8*2048*128
    u16* VT  = Kr  + (size_t)2 * NKV * SEQ * HD;     // 2*8*128*2048 (transposed)
    u16* ctx = VT  + (size_t)2 * NKV * SEQ * HD;     // 2*2048*4096

    qkv_rope_kernel<<<dim3(32, 48), dim3(256), 0, stream>>>(
        data, Wq, Wk, Wv, cosT, sinT, Qr, Kr, VT);
    attn_kernel<<<dim3(16, 64), dim3(256), 0, stream>>>(Qr, Kr, VT, ctx);
    oproj_kernel<<<dim3(32, 32), dim3(256), 0, stream>>>(ctx, Wo, out);
}
